// MLPPredictor_24601572671556
// MI455X (gfx1250) — matrix operand order repaired and checked by reading, not compiled, not run on hardware
//
#include <hip/hip_runtime.h>

typedef __attribute__((ext_vector_type(16))) _Float16 v16h;
typedef __attribute__((ext_vector_type(8)))  float    v8f;

#define IN_FEATS   128
#define KDIM       256          // 2*IN_FEATS (concat)
#define HIDDEN     256
#define NTILES     16           // HIDDEN / 16
#define KSTEPS     8            // KDIM / 32
#define WAVES_PB   8
#define MT_PER_WAVE 8
#define EDGES_PB   (WAVES_PB * MT_PER_WAVE * 16)   // 1024 edges per block
#define W1_F16_ELEMS (KDIM * HIDDEN)               // 65536 halves
#define W1_F16_BYTES (W1_F16_ELEMS * 2)            // 131072 bytes LDS

// ---------------------------------------------------------------------------
// Prep: convert W1 (f32, [K=256][N=256] row-major) into f16 fragments laid out
// exactly as the WMMA B-operand wants them, so the main kernel's LDS copy is a
// verbatim memcpy and each lane's B fragment is 32 contiguous bytes.
// Layout: frag(ntile,kstep) at ((ntile*8+kstep)*32 + lane)*16 halves.
//   lane<16 : col n = ntile*16+lane,     K = kstep*32 + j      (j=0..15)
//   lane>=16: col n = ntile*16+lane-16,  K = kstep*32 + 16 + j
// ---------------------------------------------------------------------------
__global__ void prep_w1_frags(const float* __restrict__ W1,
                              _Float16* __restrict__ wsW) {
    int i = blockIdx.x * blockDim.x + threadIdx.x;
    if (i >= W1_F16_ELEMS) return;
    int j     = i & 15;
    int lane  = (i >> 4) & 31;
    int blk   = i >> 9;          // 0..127
    int kstep = blk & 7;
    int ntile = blk >> 3;
    int n = ntile * 16 + (lane & 15);
    int k = kstep * 32 + ((lane >> 4) << 4) + j;
    wsW[i] = (_Float16)W1[k * HIDDEN + n];
}

// ---------------------------------------------------------------------------
// Main kernel: 256 threads = 8 waves. Stage W1 frags into 128KB dynamic LDS,
// then each wave processes 8 M-tiles of 16 edges with v_wmma_f32_16x16x32_f16.
// Dual accumulator chains (even/odd k-steps) to break the WMMA RAW chain.
// ---------------------------------------------------------------------------
__global__ void __launch_bounds__(256)
edge_mlp_wmma(const float* __restrict__ hsrc, const float* __restrict__ hdst,
              const int*   __restrict__ srcI, const int*   __restrict__ dstI,
              const _Float16* __restrict__ wsW,
              const float* __restrict__ b1,  const float* __restrict__ W2,
              const float* __restrict__ b2,  float* __restrict__ out,
              int nEdges)
{
    extern __shared__ _Float16 ldsW[];          // 65536 halves = 128 KB

    const int tid = threadIdx.x;

    // --- stage W1 fragments: 8192 x 16B, 32 iters x 256 threads ------------
    {
        const uint4* s = (const uint4*)wsW;
        uint4*       d = (uint4*)ldsW;
        #pragma unroll
        for (int it = 0; it < (W1_F16_BYTES / 16) / 256; ++it)
            d[tid + it * 256] = s[tid + it * 256];
    }
    __syncthreads();

    const int wave = tid >> 5;
    const int lane = tid & 31;
    const int half = lane >> 4;     // which 16-lane half
    const int lrow = lane & 15;     // edge-row within M-tile / col within N-tile
    const float b2v = b2[0];

    #pragma unroll 1
    for (int mt = 0; mt < MT_PER_WAVE; ++mt) {
        const int ebase = blockIdx.x * EDGES_PB + wave * (MT_PER_WAVE * 16) + mt * 16;
        const int e  = ebase + lrow;
        const int ec = (e < nEdges) ? e : (nEdges - 1);
        const float* srow = hsrc + (size_t)srcI[ec] * IN_FEATS;
        const float* drow = hdst + (size_t)dstI[ec] * IN_FEATS;

        // --- load A fragments for all 8 k-steps (f32 -> f16 on the fly) ----
        // lane<16 : K = k0 + {0..15};  lane>=16: K = k0 + {16..31}
        v16h a[KSTEPS];
        #pragma unroll
        for (int ks = 0; ks < KSTEPS; ++ks) {
            const int k0  = ks * 32;
            const int kr0 = k0 + half * 16;        // first run of 8
            const int kr1 = k0 + half * 16 + 8;    // second run of 8
            const float* p0 = (kr0 < IN_FEATS) ? (srow + kr0) : (drow + (kr0 - IN_FEATS));
            const float* p1 = (kr1 < IN_FEATS) ? (srow + kr1) : (drow + (kr1 - IN_FEATS));
            float4 f0 = *(const float4*)(p0);
            float4 f1 = *(const float4*)(p0 + 4);
            float4 f2 = *(const float4*)(p1);
            float4 f3 = *(const float4*)(p1 + 4);
            v16h av;
            av[0]  = (_Float16)f0.x; av[1]  = (_Float16)f0.y;
            av[2]  = (_Float16)f0.z; av[3]  = (_Float16)f0.w;
            av[4]  = (_Float16)f1.x; av[5]  = (_Float16)f1.y;
            av[6]  = (_Float16)f1.z; av[7]  = (_Float16)f1.w;
            av[8]  = (_Float16)f2.x; av[9]  = (_Float16)f2.y;
            av[10] = (_Float16)f2.z; av[11] = (_Float16)f2.w;
            av[12] = (_Float16)f3.x; av[13] = (_Float16)f3.y;
            av[14] = (_Float16)f3.z; av[15] = (_Float16)f3.w;
            a[ks] = av;
        }

        float score[8];
        #pragma unroll
        for (int r = 0; r < 8; ++r) score[r] = 0.0f;

        // --- GEMM over 16 N-tiles, fused bias+ReLU+W2 epilogue -------------
        #pragma unroll 2
        for (int nt = 0; nt < NTILES; ++nt) {
            v8f c0 = {};
            v8f c1 = {};
            #pragma unroll
            for (int ks = 0; ks < KSTEPS; ks += 2) {
                const v16h bf0 = *(const v16h*)&ldsW[((nt * KSTEPS + ks)     * 32 + lane) * 16];
                const v16h bf1 = *(const v16h*)&ldsW[((nt * KSTEPS + ks + 1) * 32 + lane) * 16];
                c0 = __builtin_amdgcn_wmma_f32_16x16x32_f16(
                         false, a[ks],     false, bf0, (short)0, c0, false, false);
                c1 = __builtin_amdgcn_wmma_f32_16x16x32_f16(
                         false, a[ks + 1], false, bf1, (short)0, c1, false, false);
            }
            const int   n  = nt * 16 + lrow;
            const float bv = b1[n];
            const float wv = W2[n];
            #pragma unroll
            for (int r = 0; r < 8; ++r) {
                const float h = fmaxf((c0[r] + c1[r]) + bv, 0.0f);
                score[r] += h * wv;
            }
        }

        // --- reduce partial scores across the 16 lanes of each half --------
        // D-layout: lanes 0-15 hold rows 0..7, lanes 16-31 hold rows 8..15;
        // xor masks 1/2/4/8 stay within each half.
        #pragma unroll
        for (int r = 0; r < 8; ++r) {
            float v = score[r];
            v += __shfl_xor(v, 1, 32);
            v += __shfl_xor(v, 2, 32);
            v += __shfl_xor(v, 4, 32);
            v += __shfl_xor(v, 8, 32);
            score[r] = v;
        }

        if (lrow == 0) {
            #pragma unroll
            for (int r = 0; r < 8; ++r) {
                const int m  = half * 8 + r;
                const int eo = ebase + m;
                if (eo < nEdges) out[eo] = score[r] + b2v;
            }
        }
    }
}

// ---------------------------------------------------------------------------
extern "C" void kernel_launch(void* const* d_in, const int* in_sizes, int n_in,
                              void* d_out, int out_size, void* d_ws, size_t ws_size,
                              hipStream_t stream) {
    const float* hsrc = (const float*)d_in[0];
    const float* hdst = (const float*)d_in[1];
    const int*   srcI = (const int*)d_in[2];
    const int*   dstI = (const int*)d_in[3];
    const float* W1   = (const float*)d_in[4];
    const float* b1   = (const float*)d_in[5];
    const float* W2   = (const float*)d_in[6];
    const float* b2   = (const float*)d_in[7];
    float*       out  = (float*)d_out;
    const int nEdges  = in_sizes[2];

    _Float16* wsW = (_Float16*)d_ws;   // 128 KB of workspace for f16 W1 frags

    prep_w1_frags<<<W1_F16_ELEMS / 256, 256, 0, stream>>>(W1, wsW);

    const int blocks = (nEdges + EDGES_PB - 1) / EDGES_PB;
    edge_mlp_wmma<<<blocks, 256, W1_F16_BYTES, stream>>>(
        hsrc, hdst, srcI, dstI, wsW, b1, W2, b2, out, nEdges);
}
